// MultiHeadAttnAlibi_36885179138294
// MI455X (gfx1250) — matrix operand order repaired and checked by reading, not compiled, not run on hardware
//
#include <hip/hip_runtime.h>
#include <math.h>

// Problem constants (reference: B=2, S=2048, D=1024, H=16, HD=64)
#define BB   2
#define SS   2048
#define DD   1024
#define HH   16
#define HDD  64

typedef __attribute__((ext_vector_type(16))) __bf16 v16bf;
typedef __attribute__((ext_vector_type(8)))  float  v8f;

union BfVec {
    v16bf          v;
    unsigned short u[16];
    uint4          q[2];
};

// f32 -> bf16 round-to-nearest-even (bit trick)
__device__ __forceinline__ unsigned short f2bf(float f) {
    unsigned int x = __float_as_uint(f);
    x += 0x7fffu + ((x >> 16) & 1u);
    return (unsigned short)(x >> 16);
}

// A-fragment (16x32 bf16): lane row M = lane&15; elements e at K = kbase+hi*16+e
// (contiguous 16). `p` = row base + kbase + hi*16 (bf16 source).
__device__ __forceinline__ v16bf load_a_frag_bf(const unsigned short* p) {
    const uint4* q = reinterpret_cast<const uint4*>(p);
    BfVec r;
    r.q[0] = q[0];   // bf16 elements p[0..7]
    r.q[1] = q[1];   // bf16 elements p[8..15]
    return r.v;
}
// B-fragment (32x16 bf16): lane col N = lane&15; elements e at K = hi*16 + e
// (contiguous 16). `p` = row base + kbase + hi*16 (bf16 source).
__device__ __forceinline__ v16bf load_b_frag_bf(const unsigned short* p) {
    const uint4* q = reinterpret_cast<const uint4*>(p);
    BfVec r;
    r.q[0] = q[0];
    r.q[1] = q[1];
    return r.v;
}

__device__ __forceinline__ v8f wmma_bf16(v16bf a, v16bf b, v8f c) {
    return __builtin_amdgcn_wmma_f32_16x16x32_bf16(false, a, false, b,
                                                   (short)0, c, false, false);
}

// 16-lane group reductions (wave32; xor masks 1,2,4,8 stay within half-wave)
__device__ __forceinline__ float rowmax16(float v) {
#pragma unroll
    for (int i = 1; i < 16; i <<= 1) v = fmaxf(v, __shfl_xor(v, i, 32));
    return v;
}
__device__ __forceinline__ float rowsum16(float v) {
#pragma unroll
    for (int i = 1; i < 16; i <<= 1) v += __shfl_xor(v, i, 32);
    return v;
}

// ---------------------------------------------------------------------------
// One-pass f32 -> bf16 (RNE) conversion; 8 elements / thread, b128 in / b128 out.
// ---------------------------------------------------------------------------
__global__ __launch_bounds__(256) void cvt_f32_bf16(const float* __restrict__ src,
                                                    unsigned short* __restrict__ dst,
                                                    int n) {
    union U8 { unsigned short u[8]; uint4 q; };
    const int i = (blockIdx.x * 256 + threadIdx.x) * 8;
    if (i >= n) return;
    const float4* s = reinterpret_cast<const float4*>(src + i);
    float4 v0 = s[0], v1 = s[1];
    U8 r;
    r.u[0] = f2bf(v0.x); r.u[1] = f2bf(v0.y); r.u[2] = f2bf(v0.z); r.u[3] = f2bf(v0.w);
    r.u[4] = f2bf(v1.x); r.u[5] = f2bf(v1.y); r.u[6] = f2bf(v1.z); r.u[7] = f2bf(v1.w);
    *reinterpret_cast<uint4*>(dst + i) = r.q;
}

// ---------------------------------------------------------------------------
// GEMM: y[m,n] = sum_k A[m,k] * W[n,k]   (A: 4096x1024 bf16, W: 1024x1024 bf16)
// mode 0: write f32 row-major (M x N)               -> out-projection
// mode 1: write bf16 (B,H,S,HD)                     -> Q, K
// mode 2: write bf16 (B,H,HD,S) (transposed)        -> V^T
// Each wave computes a 32x64 tile: 2 A-frags x 4 B-frags, 8 WMMA accumulators.
// Inner loop is pure b128 loads + WMMA (no conversion VALU).
// grid = 256 blocks x 256 threads (8 waves) = 2048 waves = (4096/32)*(1024/64)
// ---------------------------------------------------------------------------
__global__ __launch_bounds__(256) void mha_gemm_bf16(const unsigned short* __restrict__ A,
                                                     const unsigned short* __restrict__ W,
                                                     void* __restrict__ out,
                                                     int mode) {
    const int N = DD, K = DD;
    const int wave = threadIdx.x >> 5;
    const int lane = threadIdx.x & 31;
    const int lo = lane & 15, hi = lane >> 4;

    const int wid = blockIdx.x * 8 + wave;
    const int n0 = (wid & (N / 64 - 1)) * 64;  // N/64 == 16
    const int m0 = (wid >> 4) * 32;

    v8f c[2][4] = {};

    const unsigned short* a0p = A + (size_t)(m0 + lo) * K + hi * 16;
    const unsigned short* a1p = a0p + (size_t)16 * K;
    const unsigned short* bp[4];
#pragma unroll
    for (int j = 0; j < 4; j++)
        bp[j] = W + (size_t)(n0 + j * 16 + lo) * K + hi * 16;

    for (int k0 = 0; k0 < K; k0 += 32) {
        // prefetch one k-step ahead into the WGP cache (global_prefetch_b8, WGP scope)
        __builtin_prefetch(a0p + k0 + 32, 0, 3);
        __builtin_prefetch(bp[0] + k0 + 32, 0, 3);

        const v16bf a0 = load_a_frag_bf(a0p + k0);
        const v16bf a1 = load_a_frag_bf(a1p + k0);
#pragma unroll
        for (int j = 0; j < 4; j++) {
            const v16bf b = load_b_frag_bf(bp[j] + k0);
            c[0][j] = wmma_bf16(a0, b, c[0][j]);
            c[1][j] = wmma_bf16(a1, b, c[1][j]);
        }
    }

#pragma unroll
    for (int i = 0; i < 2; i++) {
#pragma unroll
        for (int j = 0; j < 4; j++) {
#pragma unroll
            for (int r = 0; r < 8; r++) {
                const int m = m0 + i * 16 + hi * 8 + r;   // D-layout: row = hi*8+r
                const int n = n0 + j * 16 + lo;           //           col = lo
                const float val = c[i][j][r];
                if (mode == 0) {
                    reinterpret_cast<float*>(out)[(size_t)m * N + n] = val;
                } else {
                    const int b = m >> 11, s = m & (SS - 1);
                    const int h = n >> 6,  hd = n & (HDD - 1);
                    size_t idx;
                    if (mode == 1)
                        idx = ((size_t)((b * HH + h) * SS + s)) * HDD + hd;  // (B,H,S,HD)
                    else
                        idx = ((size_t)((b * HH + h) * HDD + hd)) * SS + s;  // (B,H,HD,S)
                    reinterpret_cast<unsigned short*>(out)[idx] = f2bf(val);
                }
            }
        }
    }
}

// ---------------------------------------------------------------------------
// Flash attention with ALiBi + causal mask.
// Q,K: bf16 (B,H,S,HD). Vt: bf16 (B,H,HD,S). Output: bf16 row-major (B,S,D),
// consumed directly as the A operand of the Wo projection GEMM.
// One wave per 16-query tile; streams 32-key blocks with online softmax;
// exp(P) is transposed to A-fragment layout through per-wave LDS (ds_load_b128).
// grid = (16, B*H=32), block = 256 (8 waves -> 128 query tiles on grid.x).
// ---------------------------------------------------------------------------
__global__ __launch_bounds__(256) void mha_flash_alibi(
        const unsigned short* __restrict__ Q,
        const unsigned short* __restrict__ Kx,
        const unsigned short* __restrict__ Vt,
        unsigned short* __restrict__ attn) {
    __shared__ __align__(16) unsigned short plds[8][16][32];  // per-wave P tile (8 KB)

    const int wave = threadIdx.x >> 5;
    const int lane = threadIdx.x & 31;
    const int lo = lane & 15, hi = lane >> 4;

    const int bh = blockIdx.y;
    const int b = bh >> 4, h = bh & 15;
    const int q0 = (blockIdx.x * 8 + wave) * 16;

    const unsigned short* Qh = Q  + (size_t)bh * SS * HDD;
    const unsigned short* Kh = Kx + (size_t)bh * SS * HDD;
    const unsigned short* Vh = Vt + (size_t)bh * HDD * SS;

    // Q fragments for the whole row-tile (hd 0..31 and 32..63)
    const v16bf qa0 = load_a_frag_bf(Qh + (size_t)(q0 + lo) * HDD + hi * 16);
    const v16bf qa1 = load_a_frag_bf(Qh + (size_t)(q0 + lo) * HDD + 32 + hi * 16);

    const float slope = exp2f(-0.5f * (float)(h + 1));   // 2^(-(h+1)*8/H)
    const float scale = 0.125f;                          // 1/sqrt(HD)

    float mrow[8], lrow[8];
#pragma unroll
    for (int r = 0; r < 8; r++) { mrow[r] = -1e30f; lrow[r] = 0.0f; }
    v8f o0 = {}, o1 = {}, o2 = {}, o3 = {};

    for (int j0 = 0; j0 < q0 + 16; j0 += 32) {
        // scores: two 16x16 tiles (keys j0..j0+15, j0+16..j0+31), K-dim = 64
        v8f s0 = {}, s1 = {};
        s0 = wmma_bf16(qa0, load_b_frag_bf(Kh + (size_t)(j0 + lo) * HDD + hi * 16), s0);
        s0 = wmma_bf16(qa1, load_b_frag_bf(Kh + (size_t)(j0 + lo) * HDD + 32 + hi * 16), s0);
        s1 = wmma_bf16(qa0, load_b_frag_bf(Kh + (size_t)(j0 + 16 + lo) * HDD + hi * 16), s1);
        s1 = wmma_bf16(qa1, load_b_frag_bf(Kh + (size_t)(j0 + 16 + lo) * HDD + 32 + hi * 16), s1);

#pragma unroll
        for (int r = 0; r < 8; r++) {
            const int q   = q0 + hi * 8 + r;    // this row (D-layout)
            const int k0i = j0 + lo;
            const int k1i = j0 + 16 + lo;
            float v0 = (k0i <= q) ? (s0[r] * scale + slope * (float)(k0i - q)) : -1000000.0f;
            float v1 = (k1i <= q) ? (s1[r] * scale + slope * (float)(k1i - q)) : -1000000.0f;

            const float mx    = rowmax16(fmaxf(v0, v1));
            const float mnew  = fmaxf(mrow[r], mx);
            const float alpha = __expf(mrow[r] - mnew);
            const float p0 = __expf(v0 - mnew);
            const float p1 = __expf(v1 - mnew);
            lrow[r] = lrow[r] * alpha + rowsum16(p0 + p1);
            mrow[r] = mnew;
            o0[r] *= alpha; o1[r] *= alpha; o2[r] *= alpha; o3[r] *= alpha;

            // stash exp(P) in LDS (D-layout positions) for layout transpose
            plds[wave][hi * 8 + r][lo]      = f2bf(p0);
            plds[wave][hi * 8 + r][16 + lo] = f2bf(p1);
        }

        // re-read P as an A-fragment (16 queries x 32 keys); same-wave LDS is in-order
        const v16bf pf = load_a_frag_bf(&plds[wave][lo][0] + hi * 16);

        // O += P @ V   (V^T rows are contiguous in the key dimension)
        o0 = wmma_bf16(pf, load_b_frag_bf(Vh + (size_t)(0 * 16 + lo) * SS + j0 + hi * 16), o0);
        o1 = wmma_bf16(pf, load_b_frag_bf(Vh + (size_t)(1 * 16 + lo) * SS + j0 + hi * 16), o1);
        o2 = wmma_bf16(pf, load_b_frag_bf(Vh + (size_t)(2 * 16 + lo) * SS + j0 + hi * 16), o2);
        o3 = wmma_bf16(pf, load_b_frag_bf(Vh + (size_t)(3 * 16 + lo) * SS + j0 + hi * 16), o3);
    }

#pragma unroll
    for (int r = 0; r < 8; r++) {
        const float inv = 1.0f / lrow[r];
        const int q = q0 + hi * 8 + r;
        const size_t base = ((size_t)(b * SS + q)) * DD + h * HDD + lo;
        attn[base + 0]  = f2bf(o0[r] * inv);
        attn[base + 16] = f2bf(o1[r] * inv);
        attn[base + 32] = f2bf(o2[r] * inv);
        attn[base + 48] = f2bf(o3[r] * inv);
    }
}

// ---------------------------------------------------------------------------
// Host launch: convert -> QKV projections -> flash attention -> out projection.
// Workspace (bytes):
//   xb   @  0MB (8MB)   x in bf16
//   Wqb  @  8MB (2MB)   Wkb @ 10MB, Wvb @ 12MB, Wob @ 14MB
//   Qb   @ 16MB (8MB)   Kb  @ 24MB, Vtb @ 32MB
//   attb @ 40MB (8MB)   attention output, bf16 (B,S,D)
// ---------------------------------------------------------------------------
extern "C" void kernel_launch(void* const* d_in, const int* in_sizes, int n_in,
                              void* d_out, int out_size, void* d_ws, size_t ws_size,
                              hipStream_t stream) {
    (void)in_sizes; (void)n_in; (void)out_size; (void)ws_size;
    const float* x  = (const float*)d_in[0];
    // d_in[1] = causal mask (tril) -- causality is computed analytically
    const float* Wq = (const float*)d_in[2];
    const float* Wk = (const float*)d_in[3];
    const float* Wv = (const float*)d_in[4];
    const float* Wo = (const float*)d_in[5];
    float* out = (float*)d_out;

    char* ws = (char*)d_ws;
    unsigned short* xb   = (unsigned short*)(ws);
    unsigned short* Wqb  = (unsigned short*)(ws + ((size_t)8  << 20));
    unsigned short* Wkb  = (unsigned short*)(ws + ((size_t)10 << 20));
    unsigned short* Wvb  = (unsigned short*)(ws + ((size_t)12 << 20));
    unsigned short* Wob  = (unsigned short*)(ws + ((size_t)14 << 20));
    unsigned short* Qb   = (unsigned short*)(ws + ((size_t)16 << 20));
    unsigned short* Kb   = (unsigned short*)(ws + ((size_t)24 << 20));
    unsigned short* Vtb  = (unsigned short*)(ws + ((size_t)32 << 20));
    unsigned short* attb = (unsigned short*)(ws + ((size_t)40 << 20));

    const int nX = BB * SS * DD;   // 4194304
    const int nW = DD * DD;        // 1048576
    const dim3 blk(256);

    cvt_f32_bf16<<<dim3(nX / (256 * 8)), blk, 0, stream>>>(x,  xb,  nX);
    cvt_f32_bf16<<<dim3(nW / (256 * 8)), blk, 0, stream>>>(Wq, Wqb, nW);
    cvt_f32_bf16<<<dim3(nW / (256 * 8)), blk, 0, stream>>>(Wk, Wkb, nW);
    cvt_f32_bf16<<<dim3(nW / (256 * 8)), blk, 0, stream>>>(Wv, Wvb, nW);
    cvt_f32_bf16<<<dim3(nW / (256 * 8)), blk, 0, stream>>>(Wo, Wob, nW);

    const dim3 gemm_grid(256);     // 2048 waves = (4096/32) * (1024/64)
    mha_gemm_bf16<<<gemm_grid, blk, 0, stream>>>(xb, Wqb, (void*)Qb, 1);
    mha_gemm_bf16<<<gemm_grid, blk, 0, stream>>>(xb, Wkb, (void*)Kb, 1);
    mha_gemm_bf16<<<gemm_grid, blk, 0, stream>>>(xb, Wvb, (void*)Vtb, 2);

    mha_flash_alibi<<<dim3(16, BB * HH), blk, 0, stream>>>(Qb, Kb, Vtb, attb);

    mha_gemm_bf16<<<gemm_grid, blk, 0, stream>>>(attb, Wob, (void*)out, 0);
}
